// QMixer_25417616457951
// MI455X (gfx1250) — matrix operand order repaired and checked by reading, not compiled, not run on hardware
//
#include <hip/hip_runtime.h>
#include <math.h>

// ---------------- problem constants ----------------
#define TOK_TOTAL (64 * 512)   // BS*SL
#define WAVES   8
#define THREADS 256
#define BLOCKS  1024

typedef __attribute__((ext_vector_type(16))) _Float16 v16h;
typedef __attribute__((ext_vector_type(8)))  float    v8f;

// ---------------- LDS layout ----------------
// packed B fragments (halfs): frag = 32 lanes x 16 halfs = 512 halfs = 1KB
#define FH_ALLY   0        // 8 frags  (64x64)
#define FH_ENEMY  4096     // 8 frags
#define FH_Q      8192     // 8 frags
#define FH_K      12288    // 8 frags
#define FH_TRE    16384    // 8 frags
#define FH_HW1A   20480    // 24 frags (192x64)
#define FH_HW1B   32768    // 4 frags  (64x32)
#define FH_TOTAL  34816    // halfs
#define BIAS_OFF  (FH_TOTAL * 2)          // bytes (69632)
#define BIAS_FLOATS 640
#define WAVE_OFF  (BIAS_OFF + BIAS_FLOATS * 4)  // 72192
#define WAVE_BYTES 9216
#define SMEM_BYTES (WAVE_OFF + WAVES * WAVE_BYTES)  // 145920 (< 320KB WGP)

// bias float offsets inside bias region
#define B_ALLYB  0
#define B_ENEMYB 64
#define B_QB     128
#define B_KB     192
#define B_TREB   256
#define B_HW1B1  320
#define B_HW1B2  384
#define B_HWFB1  416
#define B_HWFB2  480
#define B_HB1B   512
#define B_VB1    544
#define B_VB2    576

__device__ inline v8f zero8() {
  v8f c;
#pragma unroll
  for (int i = 0; i < 8; ++i) c[i] = 0.f;
  return c;
}

__device__ inline v8f wmma16(v16h a, v16h b, v8f c) {
  // D = A(16x32 f16) * B(32x16 f16) + C(16x16 f32)
  return __builtin_amdgcn_wmma_f32_16x16x32_f16(false, a, false, b, (short)0, c,
                                                false, false);
}

__device__ inline void wavebar() { __builtin_amdgcn_wave_barrier(); }

// ---- A fragment: 16-bit A 16x32; lane: m = lane&15, hi = lane>>4
//      half h -> K = kt*32 + hi*16 + h  (contiguous)
__device__ inline v16h a_from_g32(const float* base, int ld, int row, int kt,
                                  int lane, bool valid) {
  v16h a;
  if (valid) {
    const float* p = base + row * ld + kt * 32 + ((lane >> 4) << 4);
#pragma unroll
    for (int h = 0; h < 8; ++h) a[h] = (_Float16)p[h];
#pragma unroll
    for (int h = 0; h < 8; ++h) a[8 + h] = (_Float16)p[8 + h];
  } else {
#pragma unroll
    for (int i = 0; i < 16; ++i) a[i] = (_Float16)0.f;
  }
  return a;
}

__device__ inline v16h a_from_lds(const _Float16* act, int ld, int kt, int lane,
                                  int rowsValid) {
  int m = lane & 15;
  v16h a;
  if (m < rowsValid) {
    const _Float16* p = act + m * ld + kt * 32 + ((lane >> 4) << 4);
#pragma unroll
    for (int h = 0; h < 8; ++h) a[h] = p[h];
#pragma unroll
    for (int h = 0; h < 8; ++h) a[8 + h] = p[8 + h];
  } else {
#pragma unroll
    for (int i = 0; i < 16; ++i) a[i] = (_Float16)0.f;
  }
  return a;
}

// ---- B fragment from pre-swizzled LDS: 32 contiguous bytes per lane
__device__ inline v16h b_packed(const _Float16* buf, int frag, int lane) {
  const _Float16* p = buf + ((frag << 5) + lane) * 16;
  v16h b;
#pragma unroll
  for (int i = 0; i < 16; ++i) b[i] = p[i];
  return b;
}

// ---- B fragment = transpose of a row-major [16][64] f16 tile (for q*k^T):
//      B(kk, n) = act[n][kk];  lane: n = lane&15, kk = kt*32 + hi*16 + h  (contiguous)
__device__ inline v16h b_transposed(const _Float16* act, int ld, int kt, int lane) {
  const _Float16* p = act + (lane & 15) * ld + kt * 32 + ((lane >> 4) << 4);
  v16h b;
#pragma unroll
  for (int i = 0; i < 16; ++i) b[i] = p[i];
  return b;
}

// stage f32 weight [K][N] into packed B-fragment order (f16)
__device__ inline void pack_B(const float* __restrict__ W, int K, int N,
                              _Float16* dst) {
  int nfn = N >> 4;
  int total = (K >> 5) * nfn * 512;
  for (int idx = threadIdx.x; idx < total; idx += THREADS) {
    int h = idx & 15;
    int lane = (idx >> 4) & 31;
    int f = idx >> 9;
    int nt = f % nfn, kt = f / nfn;
    int n = nt * 16 + (lane & 15);
    int kk = kt * 32 + ((lane >> 4) << 4) + h;
    dst[idx] = (_Float16)W[kk * N + n];
  }
}

__device__ inline void stage_bias(float* bf, const float* __restrict__ s, int off,
                                  int n) {
  for (int i = threadIdx.x; i < n; i += THREADS) bf[off + i] = s[i];
}

__global__ __launch_bounds__(THREADS, 1) void qmix_kernel(
    const float* __restrict__ agent_qs, const float* __restrict__ states,
    const float* __restrict__ mu, const float* __restrict__ ally_w,
    const float* __restrict__ ally_b, const float* __restrict__ enemy_w,
    const float* __restrict__ enemy_b, const float* __restrict__ q_w,
    const float* __restrict__ q_b, const float* __restrict__ k_w,
    const float* __restrict__ k_b, const float* __restrict__ hw1_w1,
    const float* __restrict__ hw1_b1, const float* __restrict__ hw1_w2,
    const float* __restrict__ hw1_b2, const float* __restrict__ hwf_w1,
    const float* __restrict__ hwf_b1, const float* __restrict__ hwf_w2,
    const float* __restrict__ hwf_b2, const float* __restrict__ hb1_w,
    const float* __restrict__ hb1_b, const float* __restrict__ v_w1,
    const float* __restrict__ v_b1, const float* __restrict__ v_w2,
    const float* __restrict__ v_b2, const float* __restrict__ tre_w,
    const float* __restrict__ tre_b, float* __restrict__ out) {
  extern __shared__ char smem[];
  _Float16* fragH = (_Float16*)smem;
  float* bf = (float*)(smem + BIAS_OFF);

  // -------- one-time block staging: weights -> f16 B fragments, biases -> f32
  pack_B(ally_w, 64, 64, fragH + FH_ALLY);
  pack_B(enemy_w, 64, 64, fragH + FH_ENEMY);
  pack_B(q_w, 64, 64, fragH + FH_Q);
  pack_B(k_w, 64, 64, fragH + FH_K);
  pack_B(tre_w, 64, 64, fragH + FH_TRE);
  pack_B(hw1_w1, 192, 64, fragH + FH_HW1A);
  pack_B(hw1_w2, 64, 32, fragH + FH_HW1B);
  stage_bias(bf, ally_b, B_ALLYB, 64);
  stage_bias(bf, enemy_b, B_ENEMYB, 64);
  stage_bias(bf, q_b, B_QB, 64);
  stage_bias(bf, k_b, B_KB, 64);
  stage_bias(bf, tre_b, B_TREB, 64);
  stage_bias(bf, hw1_b1, B_HW1B1, 64);
  stage_bias(bf, hw1_b2, B_HW1B2, 32);
  stage_bias(bf, hwf_b1, B_HWFB1, 64);
  stage_bias(bf, hwf_b2, B_HWFB2, 32);
  stage_bias(bf, hb1_b, B_HB1B, 32);
  stage_bias(bf, v_b1, B_VB1, 32);
  stage_bias(bf, v_b2, B_VB2, 1);
  __syncthreads();

  const int lane = threadIdx.x & 31;
  const int wave = threadIdx.x >> 5;
  const int m = lane & 15;
  const int hi = lane >> 4;

  char* wsb = smem + WAVE_OFF + wave * WAVE_BYTES;
  _Float16* allyH = (_Float16*)(wsb);           // [8][64] raw ally feats
  _Float16* entH = (_Float16*)(wsb + 1024);     // [16][64] embeddings
  _Float16* h1H = entH;                         // [8][64] reuse after attn
  _Float16* qH = (_Float16*)(wsb + 3072);       // [16][64]
  _Float16* kH = (_Float16*)(wsb + 5120);       // [16][64]
  _Float16* emixH = (_Float16*)(wsb + 3072);    // [8][192] reuse q/k region
  _Float16* treH = (_Float16*)(wsb + 7168);     // [8][64]
  float* mixL = (float*)(wsb + 8192);           // [128] = attn_out | mean_tre
  float* t1L = (float*)(wsb + 8704);            // [64] hwf hidden

  const int stride = gridDim.x * WAVES;
  for (int tok = blockIdx.x * WAVES + wave; tok < TOK_TOTAL; tok += stride) {
    const float* st = states + (size_t)tok * 1024;
    const float* mt = mu + (size_t)tok * 512;
    int nxt = tok + stride;
    if (nxt < TOK_TOTAL) {
      __builtin_prefetch(states + (size_t)nxt * 1024, 0, 1);  // global_prefetch
      __builtin_prefetch(mu + (size_t)nxt * 512, 0, 1);
    }

    // raw ally features (used later in ent_mix)
    for (int i = lane; i < 512; i += 32) allyH[i] = (_Float16)st[i];

    // -------- entity embed: rows0-7 ally@ally_w, rows8-15 enemy@enemy_w
#pragma unroll
    for (int nt = 0; nt < 4; ++nt) {
      v8f c = zero8();
#pragma unroll
      for (int kt = 0; kt < 2; ++kt) {
        v16h aA = a_from_g32(st, 64, m, kt, lane, m < 8);
        c = wmma16(aA, b_packed(fragH + FH_ALLY, kt * 4 + nt, lane), c);
        v16h aE = a_from_g32(st + 512, 64, m - 8, kt, lane, m >= 8);
        c = wmma16(aE, b_packed(fragH + FH_ENEMY, kt * 4 + nt, lane), c);
      }
      int n = nt * 16 + m;
      float bias = (hi == 0) ? bf[B_ALLYB + n] : bf[B_ENEMYB + n];
#pragma unroll
      for (int r = 0; r < 8; ++r)
        entH[(r + 8 * hi) * 64 + n] = (_Float16)(c[r] + bias);
    }
    wavebar();

    // -------- q / k projections
#pragma unroll
    for (int nt = 0; nt < 4; ++nt) {
      v8f cq = zero8(), ck = zero8();
#pragma unroll
      for (int kt = 0; kt < 2; ++kt) {
        v16h a = a_from_lds(entH, 64, kt, lane, 16);
        cq = wmma16(a, b_packed(fragH + FH_Q, kt * 4 + nt, lane), cq);
        ck = wmma16(a, b_packed(fragH + FH_K, kt * 4 + nt, lane), ck);
      }
      int n = nt * 16 + m;
      float qb = bf[B_QB + n], kb = bf[B_KB + n];
#pragma unroll
      for (int r = 0; r < 8; ++r) {
        qH[(r + 8 * hi) * 64 + n] = (_Float16)(cq[r] + qb);
        kH[(r + 8 * hi) * 64 + n] = (_Float16)(ck[r] + kb);
      }
    }
    wavebar();

    // -------- energy = q @ k^T  (single 16x16 tile), softmax over query axis
    v8f e = zero8();
#pragma unroll
    for (int kt = 0; kt < 2; ++kt)
      e = wmma16(a_from_lds(qH, 64, kt, lane, 16), b_transposed(kH, 64, kt, lane), e);

    // column n = lane&15 lives in lanes n and n+16 (rows 0-7 / 8-15)
    float mx = e[0];
#pragma unroll
    for (int r = 1; r < 8; ++r) mx = fmaxf(mx, e[r]);
    mx = fmaxf(mx, __shfl_xor(mx, 16, 32));
    float rs[8], ssum = 0.f;
#pragma unroll
    for (int r = 0; r < 8; ++r) {
      rs[r] = __expf(e[r] - mx);
      ssum += rs[r];
    }
    ssum += __shfl_xor(ssum, 16, 32);
    float inv = __frcp_rn(ssum);
#pragma unroll
    for (int r = 0; r < 8; ++r) rs[r] *= inv;
    // attn row-sums over columns: butterfly within each half
#pragma unroll
    for (int d = 1; d < 16; d <<= 1)
#pragma unroll
      for (int r = 0; r < 8; ++r) rs[r] += __shfl_xor(rs[r], d, 32);
    float rsO[8];
#pragma unroll
    for (int r = 0; r < 8; ++r) rsO[r] = __shfl_xor(rs[r], 16, 32);

    // attn_out[e] = (1/16) * sum_M ent[M][e] * rowsum[M]; lane -> e, e+32
    {
      int bOwn = hi << 3, bOth = (1 - hi) << 3;
      float ao0 = 0.f, ao1 = 0.f;
#pragma unroll
      for (int r = 0; r < 8; ++r) {
        float w0 = rs[r] * 0.0625f, w1 = rsO[r] * 0.0625f;
        ao0 += w0 * (float)entH[(bOwn + r) * 64 + lane] +
               w1 * (float)entH[(bOth + r) * 64 + lane];
        ao1 += w0 * (float)entH[(bOwn + r) * 64 + lane + 32] +
               w1 * (float)entH[(bOth + r) * 64 + lane + 32];
      }
      mixL[lane] = ao0;
      mixL[lane + 32] = ao1;
    }

    // -------- task repre embed: tre = mu @ tre_w + b; mean_tre = rowmean(tre)
#pragma unroll
    for (int nt = 0; nt < 4; ++nt) {
      v8f c = zero8();
#pragma unroll
      for (int kt = 0; kt < 2; ++kt)
        c = wmma16(a_from_g32(mt, 64, m, kt, lane, m < 8),
                   b_packed(fragH + FH_TRE, kt * 4 + nt, lane), c);
      int n = nt * 16 + m;
      if (hi == 0) {
        float tb = bf[B_TREB + n];
        float cs = 0.f;
#pragma unroll
        for (int r = 0; r < 8; ++r) {
          float tv = c[r] + tb;
          treH[r * 64 + n] = (_Float16)tv;
          cs += tv;
        }
        mixL[64 + n] = cs * 0.125f;  // mean over 8 agents (== mean_mu@W+b)
      }
    }
    wavebar();

    // -------- build ent_mix [8][192] = [attn_out | ally_raw | tre]
#pragma unroll
    for (int r = 0; r < 8; ++r)
      for (int c2 = lane; c2 < 192; c2 += 32) {
        float v;
        if (c2 < 64) v = mixL[c2];
        else if (c2 < 128) v = (float)allyH[r * 64 + c2 - 64];
        else v = (float)treH[r * 64 + c2 - 128];
        emixH[r * 192 + c2] = (_Float16)v;
      }
    wavebar();

    // -------- hypernet stage 1: relu(ent_mix @ hw1_w1 + b1)  -> [8][64]
#pragma unroll
    for (int nt = 0; nt < 4; ++nt) {
      v8f c = zero8();
#pragma unroll
      for (int kt = 0; kt < 6; ++kt)
        c = wmma16(a_from_lds(emixH, 192, kt, lane, 8),
                   b_packed(fragH + FH_HW1A, kt * 4 + nt, lane), c);
      int n = nt * 16 + m;
      if (hi == 0) {
        float b = bf[B_HW1B1 + n];
#pragma unroll
        for (int r = 0; r < 8; ++r)
          h1H[r * 64 + n] = (_Float16)fmaxf(c[r] + b, 0.f);
      }
    }
    wavebar();

    // -------- stage 2: w1 = |h1 @ hw1_w2 + b2|, fold agent_qs immediately
    float qs[8];
#pragma unroll
    for (int i = 0; i < 8; ++i) qs[i] = agent_qs[(size_t)tok * 8 + i];
    float hpart[2] = {0.f, 0.f};
#pragma unroll
    for (int nt = 0; nt < 2; ++nt) {
      v8f c = zero8();
#pragma unroll
      for (int kt = 0; kt < 2; ++kt)
        c = wmma16(a_from_lds(h1H, 64, kt, lane, 8),
                   b_packed(fragH + FH_HW1B, kt * 2 + nt, lane), c);
      if (hi == 0) {
        float b = bf[B_HW1B2 + nt * 16 + m];
        float hs = 0.f;
#pragma unroll
        for (int r = 0; r < 8; ++r) hs += qs[r] * fabsf(c[r] + b);
        hpart[nt] = hs;
      }
    }

    // -------- mix GEMVs (128-dim, VALU, L2-resident f32 weights)
    float b1v = bf[B_HB1B + lane];
    float t0a = bf[B_HWFB1 + lane], t0b = bf[B_HWFB1 + lane + 32];
    float t2 = bf[B_VB1 + lane];
    for (int kk = 0; kk < 128; ++kk) {
      float mv = mixL[kk];
      b1v += mv * hb1_w[kk * 32 + lane];
      t0a += mv * hwf_w1[kk * 64 + lane];
      t0b += mv * hwf_w1[kk * 64 + lane + 32];
      t2 += mv * v_w1[kk * 32 + lane];
    }
    t1L[lane] = fmaxf(t0a, 0.f);
    t1L[lane + 32] = fmaxf(t0b, 0.f);
    wavebar();
    float wfv = bf[B_HWFB2 + lane];
    for (int e2 = 0; e2 < 64; ++e2) wfv += t1L[e2] * hwf_w2[e2 * 32 + lane];
    wfv = fabsf(wfv);
    // v head: reduce t2*v_w2 over 32 lanes
    float vp = fmaxf(t2, 0.f) * v_w2[lane];
#pragma unroll
    for (int d = 1; d < 32; d <<= 1) vp += __shfl_xor(vp, d, 32);
    float vtot = vp + bf[B_VB2];

    // -------- hidden = elu(qs.w1 + b1); q_tot = hidden . wf + v
    float wf_lo = __shfl(wfv, m, 32);
    float wf_hi = __shfl(wfv, m + 16, 32);
    float b1_lo = __shfl(b1v, m, 32);
    float b1_hi = __shfl(b1v, m + 16, 32);
    float part = 0.f;
    if (hi == 0) {
      float h0 = hpart[0] + b1_lo;
      h0 = h0 > 0.f ? h0 : (__expf(h0) - 1.f);
      float h1v = hpart[1] + b1_hi;
      h1v = h1v > 0.f ? h1v : (__expf(h1v) - 1.f);
      part = h0 * wf_lo + h1v * wf_hi;
    }
#pragma unroll
    for (int d = 1; d < 32; d <<= 1) part += __shfl_xor(part, d, 32);
    if (lane == 0) out[tok] = part + vtot;
    wavebar();
  }
}

extern "C" void kernel_launch(void* const* d_in, const int* in_sizes, int n_in,
                              void* d_out, int out_size, void* d_ws,
                              size_t ws_size, hipStream_t stream) {
  const float* agent_qs = (const float*)d_in[0];
  const float* states = (const float*)d_in[1];
  const float* mu = (const float*)d_in[2];
  const float* ally_w = (const float*)d_in[3];
  const float* ally_b = (const float*)d_in[4];
  const float* enemy_w = (const float*)d_in[5];
  const float* enemy_b = (const float*)d_in[6];
  const float* q_w = (const float*)d_in[7];
  const float* q_b = (const float*)d_in[8];
  const float* k_w = (const float*)d_in[9];
  const float* k_b = (const float*)d_in[10];
  const float* hw1_w1 = (const float*)d_in[11];
  const float* hw1_b1 = (const float*)d_in[12];
  const float* hw1_w2 = (const float*)d_in[13];
  const float* hw1_b2 = (const float*)d_in[14];
  const float* hwf_w1 = (const float*)d_in[15];
  const float* hwf_b1 = (const float*)d_in[16];
  const float* hwf_w2 = (const float*)d_in[17];
  const float* hwf_b2 = (const float*)d_in[18];
  const float* hb1_w = (const float*)d_in[19];
  const float* hb1_b = (const float*)d_in[20];
  const float* v_w1 = (const float*)d_in[21];
  const float* v_b1 = (const float*)d_in[22];
  const float* v_w2 = (const float*)d_in[23];
  const float* v_b2 = (const float*)d_in[24];
  const float* tre_w = (const float*)d_in[25];
  const float* tre_b = (const float*)d_in[26];

  (void)hipFuncSetAttribute((const void*)qmix_kernel,
                            hipFuncAttributeMaxDynamicSharedMemorySize,
                            SMEM_BYTES);
  qmix_kernel<<<BLOCKS, THREADS, SMEM_BYTES, stream>>>(
      agent_qs, states, mu, ally_w, ally_b, enemy_w, enemy_b, q_w, q_b, k_w,
      k_b, hw1_w1, hw1_b1, hw1_w2, hw1_b2, hwf_w1, hwf_b1, hwf_w2, hwf_b2,
      hb1_w, hb1_b, v_w1, v_b1, v_w2, v_b2, tre_w, tre_b, (float*)d_out);
}
